// TransformerDecoderLayerWithSparseAttention_81449759801351
// MI455X (gfx1250) — matrix operand order repaired and checked by reading, not compiled, not run on hardware
//
#include <hip/hip_runtime.h>
#include <hip/hip_bf16.h>

// ---------------------------------------------------------------------------
// Types for CDNA5 WMMA (wave32): bf16 16x16x32, f32 accumulate.
// ---------------------------------------------------------------------------
typedef __bf16        v16bf __attribute__((ext_vector_type(16)));
typedef float         v8f   __attribute__((ext_vector_type(8)));
typedef unsigned int  v4u   __attribute__((ext_vector_type(4)));

union Frag16 { v4u q[2]; v16bf v; };   // 32B per-lane WMMA A/B fragment

#define B_   4
#define L_   1024
#define E_   1024
#define H_   16
#define D_   64
#define W_   128
#define BL_  4096   // B*L

__device__ __forceinline__ unsigned short f2bf(float f) {
  unsigned int u = __builtin_bit_cast(unsigned int, f);
  u += 0x7FFFu + ((u >> 16) & 1u);           // round-to-nearest-even
  return (unsigned short)(u >> 16);
}

// ---------------------------------------------------------------------------
// fp32 -> bf16 conversion (vectorized, grid-stride free: one elem4/thread)
// ---------------------------------------------------------------------------
__global__ void cvt_bf16(const float4* __restrict__ src, uint2* __restrict__ dst, int n4) {
  int i = blockIdx.x * blockDim.x + threadIdx.x;
  if (i >= n4) return;
  float4 v = src[i];
  uint2 o;
  o.x = (unsigned)f2bf(v.x) | ((unsigned)f2bf(v.y) << 16);
  o.y = (unsigned)f2bf(v.z) | ((unsigned)f2bf(v.w) << 16);
  dst[i] = o;
}

__global__ void zero_f32(float4* __restrict__ dst, int n4) {
  int i = blockIdx.x * blockDim.x + threadIdx.x;
  if (i < n4) dst[i] = make_float4(0.f, 0.f, 0.f, 0.f);
}

// ---------------------------------------------------------------------------
// Generic torch-Linear GEMM:  C[M,N] = A[M,K] (bf16) . Bw[N,K]^T (bf16) + bias
// Block tile 128x128, 8 waves of 64x32 (4x2 WMMA tiles), K-step 32.
// LDS stride 40 bf16 (80B) -> conflict-free 16-row fragment reads.
// ---------------------------------------------------------------------------
__global__ __launch_bounds__(256)
void gemm_bf16_wmma(const unsigned short* __restrict__ A, int lda,
                    const unsigned short* __restrict__ Bw, int ldb,
                    const float* __restrict__ bias,
                    float* __restrict__ C, int ldc,
                    unsigned short* __restrict__ Cb, int ldcb,
                    int K, int relu) {
  __shared__ unsigned short As[128 * 40];
  __shared__ unsigned short Bs[128 * 40];

  const int tid  = threadIdx.x;
  const int lane = tid & 31;
  const int wave = tid >> 5;
  const int wm   = wave >> 2;      // 0..1 -> wave M tile (64 rows)
  const int wn   = wave & 3;       // 0..3 -> wave N tile (32 cols)
  const int half = lane >> 4;
  const int l16  = lane & 15;

  const int m0 = blockIdx.y * 128;
  const int n0 = blockIdx.x * 128;

  // staging: each thread owns a 32B (16 bf16) chunk of the 128x32 tiles
  const int srow = tid >> 1;
  const int scol = (tid & 1) * 16;

  const unsigned short* gA = A  + (size_t)(m0 + srow) * lda + scol;
  const unsigned short* gB = Bw + (size_t)(n0 + srow) * ldb + scol;
  v4u ra0 = ((const v4u*)gA)[0], ra1 = ((const v4u*)gA)[1];
  v4u rb0 = ((const v4u*)gB)[0], rb1 = ((const v4u*)gB)[1];

  v8f acc[4][2];
#pragma unroll
  for (int i = 0; i < 4; i++)
#pragma unroll
    for (int j = 0; j < 2; j++) acc[i][j] = {};

  for (int kt = 0; kt < K; kt += 32) {
    *(v4u*)&As[srow * 40 + scol]     = ra0;
    *(v4u*)&As[srow * 40 + scol + 8] = ra1;
    *(v4u*)&Bs[srow * 40 + scol]     = rb0;
    *(v4u*)&Bs[srow * 40 + scol + 8] = rb1;
    __syncthreads();

    if (kt + 32 < K) {  // software-pipeline next K-tile global loads
      const unsigned short* nA = A  + (size_t)(m0 + srow) * lda + (kt + 32) + scol;
      const unsigned short* nB = Bw + (size_t)(n0 + srow) * ldb + (kt + 32) + scol;
      ra0 = ((const v4u*)nA)[0]; ra1 = ((const v4u*)nA)[1];
      rb0 = ((const v4u*)nB)[0]; rb1 = ((const v4u*)nB)[1];
    }

    Frag16 af[4];
#pragma unroll
    for (int mi = 0; mi < 4; mi++) {
      int row = wm * 64 + mi * 16 + l16;            // A: lane row = l&15
      af[mi].q[0] = *(const v4u*)&As[row * 40 + half * 16];       // K: half*16..+7
      af[mi].q[1] = *(const v4u*)&As[row * 40 + half * 16 + 8];   // K: half*16+8..+15
    }
#pragma unroll
    for (int ni = 0; ni < 2; ni++) {
      int row = wn * 32 + ni * 16 + l16;            // B: lane col = l&15
      Frag16 bf;
      bf.q[0] = *(const v4u*)&Bs[row * 40 + half * 16];           // K: half*16..+15
      bf.q[1] = *(const v4u*)&Bs[row * 40 + half * 16 + 8];
#pragma unroll
      for (int mi = 0; mi < 4; mi++)
        acc[mi][ni] = __builtin_amdgcn_wmma_f32_16x16x32_bf16(
            false, af[mi].v, false, bf.v, (short)0, acc[mi][ni], false, false);
    }
    __syncthreads();
  }

#pragma unroll
  for (int ni = 0; ni < 2; ni++) {
    int col = n0 + wn * 32 + ni * 16 + l16;
    float bv = bias ? bias[col] : 0.0f;
#pragma unroll
    for (int mi = 0; mi < 4; mi++) {
      int rbase = m0 + wm * 64 + mi * 16 + half * 8;  // D layout: row = r + half*8
#pragma unroll
      for (int r = 0; r < 8; r++) {
        float v = acc[mi][ni][r] + bv;
        if (relu) v = fmaxf(v, 0.0f);
        if (C)  C [(size_t)(rbase + r) * ldc  + col] = v;
        if (Cb) Cb[(size_t)(rbase + r) * ldcb + col] = f2bf(v);
      }
    }
  }
}

// ---------------------------------------------------------------------------
// qkv [B,L,3E] f32 -> Q[b,h,l,d], K[b,h,l,d] bf16 and V transposed [b,h,d,l]
// (V^T makes the P.V B-fragment a contiguous 32B load per lane)
// ---------------------------------------------------------------------------
__global__ void split_qkv(const float* __restrict__ qkv,
                          unsigned short* __restrict__ qb,
                          unsigned short* __restrict__ kb,
                          unsigned short* __restrict__ vt) {
  int idx = blockIdx.x * blockDim.x + threadIdx.x;    // over B*H*L*D
  if (idx >= B_ * H_ * L_ * D_) return;
  int d = idx & 63;
  int i = (idx >> 6) & 1023;
  int h = (idx >> 16) & 15;
  int b = idx >> 20;
  const float* base = qkv + ((size_t)(b * L_ + i)) * 3 * E_ + h * D_ + d;
  qb[idx] = f2bf(base[0]);
  kb[idx] = f2bf(base[E_]);
  vt[(((size_t)b * H_ + h) * D_ + d) * L_ + i] = f2bf(base[2 * E_]);
}

// ---------------------------------------------------------------------------
// Banded attention: one wave per (b, h, 16-query tile). QK^T + mask + exact
// softmax + P.V all via bf16 WMMA; head-mean attn probs atomically added.
// ---------------------------------------------------------------------------
__global__ __launch_bounds__(32)
void attn_banded(const unsigned short* __restrict__ qb,
                 const unsigned short* __restrict__ kb,
                 const unsigned short* __restrict__ vt,
                 unsigned short* __restrict__ ctxb,
                 float* __restrict__ attnw) {
  __shared__ float          S[16 * 276];   // masked scores (stride 276: conflict-free)
  __shared__ unsigned short P[16 * 304];   // bf16 probs, padded for K=32 steps

  const int lane = threadIdx.x;
  const int half = lane >> 4;
  const int l16  = lane & 15;
  const int qt = blockIdx.x, h = blockIdx.y, b = blockIdx.z;
  const int i0  = qt * 16;
  const int jlo = max(0, i0 - W_);
  const int jhi = min(L_, i0 + 16 + W_);
  const int nt  = (jhi - jlo) >> 4;        // #16-key tiles in band (<= 17)
  const int bh  = b * H_ + h;

  const unsigned short* Q  = qb + (size_t)bh * L_ * D_;
  const unsigned short* Kp = kb + (size_t)bh * L_ * D_;
  const unsigned short* Vp = vt + (size_t)bh * D_ * L_;

  Frag16 aq[2];
  {
    const unsigned short* qrow = Q + (size_t)(i0 + l16) * D_;
#pragma unroll
    for (int c = 0; c < 2; c++) {
      aq[c].q[0] = *(const v4u*)(qrow + c * 32 + half * 16);
      aq[c].q[1] = *(const v4u*)(qrow + c * 32 + half * 16 + 8);
    }
  }

  for (int t = 0; t < nt; t++) {
    int j0 = jlo + t * 16;
    const unsigned short* krow = Kp + (size_t)(j0 + l16) * D_;
    v8f s = {};
#pragma unroll
    for (int c = 0; c < 2; c++) {
      Frag16 bk;
      bk.q[0] = *(const v4u*)(krow + c * 32 + half * 16);
      bk.q[1] = *(const v4u*)(krow + c * 32 + half * 16 + 8);
      s = __builtin_amdgcn_wmma_f32_16x16x32_bf16(
          false, aq[c].v, false, bk.v, (short)0, s, false, false);
    }
    int col = j0 + l16;
#pragma unroll
    for (int r = 0; r < 8; r++) {
      int row = i0 + half * 8 + r;
      float v = s[r] * 0.125f;                     // 1/sqrt(64)
      int dd = row - col;
      if (dd > W_ || dd < -W_) v = -1e30f;         // banded mask
      S[(half * 8 + r) * 276 + t * 16 + l16] = v;
    }
  }
  __syncthreads();

  const int ncol = nt * 16;
  const int ntk  = (nt + 1) >> 1;                  // #K=32 steps for P.V
  if (lane < 16) {                                 // lane owns query row i0+lane
    float* srow = &S[lane * 276];
    float mx = -1e30f;
    for (int c = 0; c < ncol; c++) mx = fmaxf(mx, srow[c]);
    float sum = 0.f;
    for (int c = 0; c < ncol; c++) { float e = __expf(srow[c] - mx); srow[c] = e; sum += e; }
    float inv = 1.f / sum;
    unsigned short* prow = &P[lane * 304];
    float* aw = attnw + ((size_t)b * L_ + (i0 + lane)) * L_ + jlo;
    for (int c = 0; c < ncol; c++) {
      float p = srow[c] * inv;
      prow[c] = f2bf(p);
      atomicAdd(&aw[c], p * (1.0f / H_));          // mean over heads
    }
    for (int c = ncol; c < ntk * 32; c++) prow[c] = 0;
  }
  __syncthreads();

  v8f cacc[4]; 
#pragma unroll
  for (int dt = 0; dt < 4; dt++) cacc[dt] = {};
  for (int kt = 0; kt < ntk; kt++) {
    Frag16 ap;                                     // P as A-fragment (16x32)
    ap.q[0] = *(const v4u*)&P[l16 * 304 + kt * 32 + half * 16];
    ap.q[1] = *(const v4u*)&P[l16 * 304 + kt * 32 + half * 16 + 8];
#pragma unroll
    for (int dt = 0; dt < 4; dt++) {
      Frag16 bv;                                   // V^T row d contiguous in j
      const unsigned short* vrow = Vp + (size_t)(dt * 16 + l16) * L_ + jlo + kt * 32 + half * 16;
      bv.q[0] = ((const v4u*)vrow)[0];
      bv.q[1] = ((const v4u*)vrow)[1];
      cacc[dt] = __builtin_amdgcn_wmma_f32_16x16x32_bf16(
          false, ap.v, false, bv.v, (short)0, cacc[dt], false, false);
    }
  }

#pragma unroll
  for (int dt = 0; dt < 4; dt++) {
    int col = h * D_ + dt * 16 + l16;
#pragma unroll
    for (int r = 0; r < 8; r++) {
      int row = i0 + half * 8 + r;
      ctxb[((size_t)b * L_ + row) * E_ + col] = f2bf(cacc[dt][r]);
    }
  }
}

// ---------------------------------------------------------------------------
// Fused residual + LayerNorm over E=1024; 256 threads = one row (float4/thread)
// ---------------------------------------------------------------------------
__global__ __launch_bounds__(256)
void add_layernorm(const float* __restrict__ X, const float* __restrict__ Y,
                   const float* __restrict__ g, const float* __restrict__ bta,
                   float* __restrict__ out, unsigned short* __restrict__ outb) {
  __shared__ float rs[256], rq[256];
  int row = blockIdx.x, tid = threadIdx.x;
  float4 xv = ((const float4*)(X + (size_t)row * E_))[tid];
  float4 yv = ((const float4*)(Y + (size_t)row * E_))[tid];
  float4 s = make_float4(xv.x + yv.x, xv.y + yv.y, xv.z + yv.z, xv.w + yv.w);
  rs[tid] = s.x + s.y + s.z + s.w;
  rq[tid] = s.x * s.x + s.y * s.y + s.z * s.z + s.w * s.w;
  __syncthreads();
  for (int st = 128; st > 0; st >>= 1) {
    if (tid < st) { rs[tid] += rs[tid + st]; rq[tid] += rq[tid + st]; }
    __syncthreads();
  }
  float mu   = rs[0] * (1.0f / E_);
  float var  = rq[0] * (1.0f / E_) - mu * mu;
  float rinv = rsqrtf(var + 1e-5f);
  float4 gv = ((const float4*)g)[tid];
  float4 bv = ((const float4*)bta)[tid];
  float4 o = make_float4((s.x - mu) * rinv * gv.x + bv.x,
                         (s.y - mu) * rinv * gv.y + bv.y,
                         (s.z - mu) * rinv * gv.z + bv.z,
                         (s.w - mu) * rinv * gv.w + bv.w);
  ((float4*)(out + (size_t)row * E_))[tid] = o;
  if (outb) {
    uint2 ob;
    ob.x = (unsigned)f2bf(o.x) | ((unsigned)f2bf(o.y) << 16);
    ob.y = (unsigned)f2bf(o.z) | ((unsigned)f2bf(o.w) << 16);
    ((uint2*)(outb + (size_t)row * E_))[tid] = ob;
  }
}

// ---------------------------------------------------------------------------
extern "C" void kernel_launch(void* const* d_in, const int* in_sizes, int n_in,
                              void* d_out, int out_size, void* d_ws, size_t ws_size,
                              hipStream_t stream) {
  (void)in_sizes; (void)n_in; (void)out_size; (void)ws_size;
  const float* x    = (const float*)d_in[0];
  const float* wqkv = (const float*)d_in[1];
  const float* bqkv = (const float*)d_in[2];
  const float* wo   = (const float*)d_in[3];
  const float* bo   = (const float*)d_in[4];
  const float* g1   = (const float*)d_in[5];
  const float* be1  = (const float*)d_in[6];
  const float* w1   = (const float*)d_in[7];
  const float* b1   = (const float*)d_in[8];
  const float* w2   = (const float*)d_in[9];
  const float* b2   = (const float*)d_in[10];
  const float* g2   = (const float*)d_in[11];
  const float* be2  = (const float*)d_in[12];

  char* ws = (char*)d_ws;
  size_t off = 0;
  auto take = [&](size_t bytes) -> char* {
    char* p = ws + off; off += (bytes + 255) & ~(size_t)255; return p;
  };

  unsigned short* xb    = (unsigned short*)take((size_t)BL_ * E_ * 2);
  unsigned short* wqkvb = (unsigned short*)take((size_t)3 * E_ * E_ * 2);
  unsigned short* wob   = (unsigned short*)take((size_t)E_ * E_ * 2);
  unsigned short* w1b   = (unsigned short*)take((size_t)4 * E_ * E_ * 2);
  unsigned short* w2b   = (unsigned short*)take((size_t)4 * E_ * E_ * 2);
  float*          qkv   = (float*)take((size_t)BL_ * 3 * E_ * 4);
  unsigned short* qbuf  = (unsigned short*)take((size_t)B_ * H_ * L_ * D_ * 2);
  unsigned short* kbuf  = (unsigned short*)take((size_t)B_ * H_ * L_ * D_ * 2);
  unsigned short* vt    = (unsigned short*)take((size_t)B_ * H_ * L_ * D_ * 2 + 1024); // slack for band overrun
  unsigned short* ctxb  = (unsigned short*)take((size_t)BL_ * E_ * 2);
  float*          aout  = (float*)take((size_t)BL_ * E_ * 4);
  float*          hbuf  = (float*)take((size_t)BL_ * E_ * 4);
  unsigned short* hb    = (unsigned short*)take((size_t)BL_ * E_ * 2);
  unsigned short* ffb   = (unsigned short*)take((size_t)BL_ * 4 * E_ * 2);
  float*          ff2   = (float*)take((size_t)BL_ * E_ * 4);

  float* out_f = (float*)d_out;
  float* attnw = out_f + (size_t)B_ * L_ * E_;   // second tuple element [B,L,L]

  auto cvt = [&](const float* s, unsigned short* d, size_t n) {
    int n4 = (int)(n / 4);
    cvt_bf16<<<dim3((n4 + 255) / 256), dim3(256), 0, stream>>>((const float4*)s, (uint2*)d, n4);
  };
  cvt(x,    xb,    (size_t)BL_ * E_);
  cvt(wqkv, wqkvb, (size_t)3 * E_ * E_);
  cvt(wo,   wob,   (size_t)E_ * E_);
  cvt(w1,   w1b,   (size_t)4 * E_ * E_);
  cvt(w2,   w2b,   (size_t)4 * E_ * E_);

  { int n4 = B_ * L_ * L_ / 4;   // attn_weights zero-fill (outside-band stays 0)
    zero_f32<<<dim3((n4 + 255) / 256), dim3(256), 0, stream>>>((float4*)attnw, n4); }

  // qkv = x . Wqkv^T + b   [4096, 3072]
  gemm_bf16_wmma<<<dim3(3 * E_ / 128, BL_ / 128), 256, 0, stream>>>(
      xb, E_, wqkvb, E_, bqkv, qkv, 3 * E_, (unsigned short*)nullptr, 0, E_, 0);

  split_qkv<<<dim3(B_ * H_ * L_ * D_ / 256), 256, 0, stream>>>(qkv, qbuf, kbuf, vt);

  attn_banded<<<dim3(L_ / 16, H_, B_), 32, 0, stream>>>(qbuf, kbuf, vt, ctxb, attnw);

  // attn_out = ctx . Wo^T + b   [4096, 1024]
  gemm_bf16_wmma<<<dim3(E_ / 128, BL_ / 128), 256, 0, stream>>>(
      ctxb, E_, wob, E_, bo, aout, E_, (unsigned short*)nullptr, 0, E_, 0);

  // h = LN(x + attn_out)
  add_layernorm<<<dim3(BL_), 256, 0, stream>>>(x, aout, g1, be1, hbuf, hb);

  // ff = relu(h . W1^T + b1)   [4096, 4096], bf16 out only
  gemm_bf16_wmma<<<dim3(4 * E_ / 128, BL_ / 128), 256, 0, stream>>>(
      hb, E_, w1b, E_, b1, (float*)nullptr, 0, ffb, 4 * E_, E_, 1);

  // ff2 = ff . W2^T + b2   [4096, 1024]
  gemm_bf16_wmma<<<dim3(E_ / 128, BL_ / 128), 256, 0, stream>>>(
      ffb, 4 * E_, w2b, 4 * E_, b2, ff2, E_, (unsigned short*)nullptr, 0, 4 * E_, 0);

  // out = LN(h + ff2)
  add_layernorm<<<dim3(BL_), 256, 0, stream>>>(hbuf, ff2, g2, be2, out_f, (unsigned short*)nullptr);
}
